// FlashCrossAttentionV2_3_80728205296265
// MI455X (gfx1250) — matrix operand order repaired and checked by reading, not compiled, not run on hardware
//
#include <hip/hip_runtime.h>

typedef __attribute__((ext_vector_type(16))) _Float16 v16h;
typedef __attribute__((ext_vector_type(8)))  _Float16 v8h;
typedef __attribute__((ext_vector_type(8)))  float    v8f;
typedef __attribute__((ext_vector_type(4)))  float    v4f;
typedef __attribute__((ext_vector_type(4)))  int      v4i;

#define SQ     2048
#define SK     2048
#define NH     16
#define DH     64
#define QTILE  128      // q rows per workgroup (8 waves x 16 rows)
#define KTILE  64       // keys per iteration
#define WLEFT  256
#define WRIGHT 256
#define LSTR   72       // padded LDS row stride in halves (144B, 16B aligned)
#define KVROW  (2 * NH * DH)   // float stride between consecutive seq positions in kv

__device__ __forceinline__ v8f wmma_f16(v16h a, v16h b, v8f c) {
    return __builtin_amdgcn_wmma_f32_16x16x32_f16(
        false, a, false, b, (short)0, c, false, false);
}

// A-fragment (16x32 f16) chunk loader from an LDS row:
//   halves 0..7  <- K = base + 8*hih + 0..7
//   halves 8..15 <- K = base + 16 + 8*hih + 0..7
__device__ __forceinline__ v16h ldsAfrag(const _Float16* rowp, int base, int hih) {
    v8h lo = *(const v8h*)(rowp + base + 8 * hih);
    v8h hi = *(const v8h*)(rowp + base + 16 + 8 * hih);
    return __builtin_shufflevector(lo, hi, 0,1,2,3,4,5,6,7,8,9,10,11,12,13,14,15);
}

__global__ __launch_bounds__(256, 1)
void fa_window_kernel(const float* __restrict__ q,
                      const float* __restrict__ kv,
                      float* __restrict__ out)
{
    __shared__ __attribute__((aligned(16))) _Float16 sK [KTILE * LSTR];   // [key][d]
    __shared__ __attribute__((aligned(16))) _Float16 sVt[DH    * LSTR];   // [d][key] (transposed)

    const int tid  = threadIdx.x;
    const int lane = tid & 31;
    const int wv   = tid >> 5;        // wave 0..7
    const int hih  = lane >> 4;       // lane half
    const int col  = lane & 15;

    const int qtPerBH = SQ / QTILE;   // 16
    const int blk = blockIdx.x;
    const int b   = blk / (NH * qtPerBH);
    const int h   = (blk / qtPerBH) % NH;
    const int q0  = (blk % qtPerBH) * QTILE;

    // ---- Q as B-operand of S^T = K * Q^T ----
    // B layout (32x16): N = col = q row within wave tile; halves 0..7 <-> d = f*32 + 8*hih + h, halves 8..15 <-> d = f*32 + 16 + 8*hih + (h-8)
    const int    qm   = q0 + wv * 16 + col;            // this lane's query row
    const float* qrow = q + ((size_t)(b * SQ + qm) * NH + h) * DH;
    v16h qb[2];
#pragma unroll
    for (int f = 0; f < 2; ++f) {
        const int d0 = f * 32 + 8 * hih;
        v4f x0 = *(const v4f*)(qrow + d0);
        v4f x1 = *(const v4f*)(qrow + d0 + 4);
        v4f x2 = *(const v4f*)(qrow + d0 + 16);
        v4f x3 = *(const v4f*)(qrow + d0 + 20);
        v16h a;
#pragma unroll
        for (int j = 0; j < 4; ++j) {
            a[j]      = (_Float16)(x0[j] * 0.125f);    // fold 1/sqrt(D)
            a[4 + j]  = (_Float16)(x1[j] * 0.125f);
            a[8 + j]  = (_Float16)(x2[j] * 0.125f);
            a[12 + j] = (_Float16)(x3[j] * 0.125f);
        }
        qb[f] = a;
    }

    // ---- per-lane flash state: ONE query row per lane ----
    v8f  acc[4] = {};                 // O^T tiles: row d = dt*16 + r + 8*hih, col m = qm
    float mrun = -1e30f, lrun = 0.0f;

    int klo = q0 - WLEFT;            if (klo < 0)  klo = 0;
    int khi = q0 + QTILE + WRIGHT;   if (khi > SK) khi = SK;   // exclusive, 64-aligned

    // ---- register double-buffer of the K/V tile (global latency overlaps compute) ----
    v4f kreg[4], vreg[4];
    {
        const float* kg = kv + (((size_t)(b * SK + klo) * 2 + 0) * NH + h) * DH;
        const float* vg = kv + (((size_t)(b * SK + klo) * 2 + 1) * NH + h) * DH;
#pragma unroll
        for (int it = 0; it < 4; ++it) {
            int i = tid + it * 256, krow = i >> 4, c4 = (i & 15) * 4;
            kreg[it] = *(const v4f*)(kg + (size_t)krow * KVROW + c4);
            vreg[it] = *(const v4f*)(vg + (size_t)krow * KVROW + c4);
        }
    }

    const int wq0 = q0 + wv * 16;     // wave's min / max query row
    const int wq15 = wq0 + 15;

    for (int kb = klo; kb < khi; kb += KTILE) {
        // ---- publish prefetched tile to LDS (f16; V transposed) ----
#pragma unroll
        for (int it = 0; it < 4; ++it) {
            int i = tid + it * 256, krow = i >> 4, c4 = (i & 15) * 4;
            _Float16* dk = &sK[krow * LSTR + c4];
            dk[0] = (_Float16)kreg[it][0]; dk[1] = (_Float16)kreg[it][1];
            dk[2] = (_Float16)kreg[it][2]; dk[3] = (_Float16)kreg[it][3];
            sVt[(c4 + 0) * LSTR + krow] = (_Float16)vreg[it][0];
            sVt[(c4 + 1) * LSTR + krow] = (_Float16)vreg[it][1];
            sVt[(c4 + 2) * LSTR + krow] = (_Float16)vreg[it][2];
            sVt[(c4 + 3) * LSTR + krow] = (_Float16)vreg[it][3];
        }
        __syncthreads();

        // ---- issue next tile's global loads now; consumed next iteration ----
        const int kn = kb + KTILE;
        if (kn < khi) {
            const float* kg = kv + (((size_t)(b * SK + kn) * 2 + 0) * NH + h) * DH;
            const float* vg = kv + (((size_t)(b * SK + kn) * 2 + 1) * NH + h) * DH;
#pragma unroll
            for (int it = 0; it < 4; ++it) {
                int i = tid + it * 256, krow = i >> 4, c4 = (i & 15) * 4;
                kreg[it] = *(const v4f*)(kg + (size_t)krow * KVROW + c4);
                vreg[it] = *(const v4f*)(vg + (size_t)krow * KVROW + c4);
            }
        }

        // ---- S^T = K * Q^T : 4 key-subtiles, 2 WMMAs each (K=32 over d) ----
        v8f s4[4];
#pragma unroll
        for (int nt = 0; nt < 4; ++nt) {
            const _Float16* krow = &sK[(nt * 16 + col) * LSTR];
            v16h ka0 = ldsAfrag(krow, 0, hih);
            v16h ka1 = ldsAfrag(krow, 32, hih);
            v8f c = {};
            c = wmma_f16(ka0, qb[0], c);
            c = wmma_f16(ka1, qb[1], c);
            s4[nt] = c;          // element r: key n = kb + nt*16 + r + 8*hih, row m = qm
        }

        // ---- window mask, only on boundary tiles (wave-uniform skip) ----
        const bool edge = (kb < wq15 - WLEFT) || (kb + KTILE - 1 > wq0 + WRIGHT);
        if (edge) {
            const int nlo = qm - WLEFT, nhi = qm + WRIGHT;
#pragma unroll
            for (int nt = 0; nt < 4; ++nt)
#pragma unroll
                for (int r = 0; r < 8; ++r) {
                    const int n = kb + nt * 16 + r + 8 * hih;
                    if (n < nlo || n > nhi) s4[nt][r] = -1e30f;
                }
        }

        // ---- online softmax: in-lane reduce + single cross-half shuffle ----
        float t = -1e30f;
#pragma unroll
        for (int nt = 0; nt < 4; ++nt)
#pragma unroll
            for (int r = 0; r < 8; ++r) t = fmaxf(t, s4[nt][r]);
        t = fmaxf(t, __shfl_xor(t, 16, 32));

        const float mn    = fmaxf(mrun, t);
        const float alpha = __expf(mrun - mn);   // 0 when old max was -1e30 (resets garbage)
        mrun = mn;
        lrun *= alpha;
#pragma unroll
        for (int dt = 0; dt < 4; ++dt)
#pragma unroll
            for (int r = 0; r < 8; ++r) acc[dt][r] *= alpha;

        float rs = 0.0f;
        v8h pk[4];
#pragma unroll
        for (int nt = 0; nt < 4; ++nt)
#pragma unroll
            for (int r = 0; r < 8; ++r) {
                float p = __expf(s4[nt][r] - mn);
                rs += p;
                pk[nt][r] = (_Float16)p;
            }
        rs += __shfl_xor(rs, 16, 32);
        lrun += rs;

        // ---- O^T += V^T * P^T : build P^T B-frag via one half-wave swap ----
#pragma unroll
        for (int kc = 0; kc < 2; ++kc) {
            v8h give = hih ? pk[2 * kc] : pk[2 * kc + 1];
            v4i gi = __builtin_bit_cast(v4i, give);
            gi[0] = __shfl_xor(gi[0], 16, 32);
            gi[1] = __shfl_xor(gi[1], 16, 32);
            gi[2] = __shfl_xor(gi[2], 16, 32);
            gi[3] = __shfl_xor(gi[3], 16, 32);
            v8h recv = __builtin_bit_cast(v8h, gi);
            v8h loH = hih ? recv            : pk[2 * kc];
            v8h hiH = hih ? pk[2 * kc + 1]  : recv;
            v16h pb = __builtin_shufflevector(loH, hiH, 0,1,2,3,4,5,6,7,8,9,10,11,12,13,14,15);
#pragma unroll
            for (int dt = 0; dt < 4; ++dt) {
                const _Float16* vrow = &sVt[(dt * 16 + col) * LSTR];
                v16h va = __builtin_shufflevector(*(const v8h*)(vrow + kc * 32 + 16 * hih), *(const v8h*)(vrow + kc * 32 + 16 * hih + 8), 0,1,2,3,4,5,6,7,8,9,10,11,12,13,14,15);
                acc[dt] = wmma_f16(va, pb, acc[dt]);
            }
        }
        __syncthreads();   // tile fully consumed before next LDS overwrite
    }

    // ---- epilogue: normalize, coalesced b128 stores ----
    const float inv = 1.0f / lrun;
    float* orow = out + ((size_t)(b * SQ + qm) * NH + h) * DH;
#pragma unroll
    for (int dt = 0; dt < 4; ++dt) {
        v4f o0, o1;
#pragma unroll
        for (int j = 0; j < 4; ++j) { o0[j] = acc[dt][j] * inv; o1[j] = acc[dt][4 + j] * inv; }
        *(v4f*)(orow + dt * 16 + 8 * hih)     = o0;
        *(v4f*)(orow + dt * 16 + 8 * hih + 4) = o1;
    }
}

extern "C" void kernel_launch(void* const* d_in, const int* in_sizes, int n_in,
                              void* d_out, int out_size, void* d_ws, size_t ws_size,
                              hipStream_t stream) {
    (void)in_sizes; (void)n_in; (void)out_size; (void)d_ws; (void)ws_size;
    const float* q  = (const float*)d_in[0];
    const float* kv = (const float*)d_in[1];
    float* out = (float*)d_out;
    // setup_inputs(): causal=0, use_window_mask=1 -> windowed path hardcoded.
    dim3 grid(2 * NH * (SQ / QTILE));   // 512 workgroups
    fa_window_kernel<<<grid, 256, 0, stream>>>(q, kv, out);
}
